// SkipAttention_4578435138068
// MI455X (gfx1250) — matrix operand order repaired and checked by reading, not compiled, not run on hardware
//
#include <hip/hip_runtime.h>
#include <hip/hip_bf16.h>

typedef __attribute__((ext_vector_type(16))) __bf16 v16bf;
typedef __attribute__((ext_vector_type(8)))  float  v8f;

union V16 { v16bf v; uint4 q[2]; };

#define NB   8
#define LSEQ 4096
#define HDIM 48
#define DPAD 64
#define CDIM 384

// ---- workspace layout (bytes, all 256-aligned) ----
static constexpr size_t OFF_Y32  = 0;                                   // f32 (b,pos,384)
static constexpr size_t SZ_Y32   = (size_t)NB*LSEQ*CDIM*4;
static constexpr size_t OFF_YBF  = OFF_Y32 + SZ_Y32;                    // bf16 (b,pos,384)
static constexpr size_t SZ_YBF   = (size_t)NB*LSEQ*CDIM*2;
static constexpr size_t OFF_WQKV = OFF_YBF + SZ_YBF;                    // bf16 [3][48*384]
static constexpr size_t SZ_WQKV  = (size_t)3*HDIM*CDIM*2;
static constexpr size_t OFF_Q    = OFF_WQKV + SZ_WQKV;                  // bf16 (b,pos,64)
static constexpr size_t SZ_QK    = (size_t)NB*LSEQ*DPAD*2;
static constexpr size_t OFF_K    = OFF_Q + SZ_QK;                       // bf16 (b,pos,64)
static constexpr size_t OFF_V    = OFF_K + SZ_QK;                       // bf16 (b,48,pos)
static constexpr size_t SZ_V     = (size_t)NB*HDIM*LSEQ*2;
static constexpr size_t OFF_AO   = OFF_V + SZ_V;                        // f32 (b,pos,48)

__device__ inline v8f vzero() {
  v8f z = {0.f,0.f,0.f,0.f,0.f,0.f,0.f,0.f};
  return z;
}
__device__ inline v8f wmma_bf16(v16bf a, v16bf b, v8f c) {
  return __builtin_amdgcn_wmma_f32_16x16x32_bf16(false, a, false, b, (short)0, c, false, false);
}

// ---------------- kernel 0: convert qkv weights to bf16 ----------------
__global__ void cvt_w_kernel(const float* __restrict__ wq, const float* __restrict__ wk,
                             const float* __restrict__ wv, __bf16* __restrict__ dst) {
  int i = blockIdx.x * 256 + threadIdx.x;
  int n = HDIM * CDIM;
  if (i < n) {
    dst[i]       = (__bf16)wq[i];
    dst[n + i]   = (__bf16)wk[i];
    dst[2*n + i] = (__bf16)wv[i];
  }
}

// ---------------- kernel 1: patch ops + concat + LayerNorm ----------------
// one block (96 threads) per output position; thread o owns channels {o, 96+o, 192+o, 288+o}
__global__ __launch_bounds__(96)
void fuse_ln_kernel(const float* __restrict__ x0, const float* __restrict__ x1,
                    const float* __restrict__ x2, const float* __restrict__ x3,
                    const float* __restrict__ wm, const float* __restrict__ bm,
                    const float* __restrict__ we2, const float* __restrict__ be2,
                    const float* __restrict__ we4, const float* __restrict__ be4,
                    const float* __restrict__ lnw, const float* __restrict__ lnb,
                    float* __restrict__ y32, __bf16* __restrict__ ybf) {
  const int W = blockIdx.x, H = blockIdx.y, b = blockIdx.z;
  const int o = threadIdx.x;
  __shared__ float xm[192], xe2[48], xe4[24];
  __shared__ float red[96], red2[96];
  __shared__ float stat[2];

  for (int t = o; t < 192; t += 96) {
    int c = t >> 2, i = (t >> 1) & 1, j = t & 1;
    xm[t] = x0[(((size_t)b*48 + c)*128 + (2*H + i))*128 + (2*W + j)];
  }
  if (o < 48)
    xe2[o] = x2[(((size_t)b*192 + (o*4 + (H&1)*2 + (W&1)))*32 + (H>>1))*32 + (W>>1)];
  if (o < 24)
    xe4[o] = x3[(((size_t)b*384 + (o*16 + (H&3)*4 + (W&3)))*16 + (H>>2))*16 + (W>>2)];
  __syncthreads();

  float a0 = bm[o];
  #pragma unroll 4
  for (int i = 0; i < 192; ++i) a0 = fmaf(xm[i], wm[i*96 + o], a0);
  float a2 = be2[o];
  #pragma unroll 4
  for (int i = 0; i < 48; ++i) a2 = fmaf(xe2[i], we2[i*96 + o], a2);
  float a3 = be4[o];
  #pragma unroll 4
  for (int i = 0; i < 24; ++i) a3 = fmaf(xe4[i], we4[i*96 + o], a3);
  float a1 = x1[(((size_t)b*96 + o)*64 + H)*64 + W];

  float v0 = a0 + a1, v1 = a1 + a2, v2 = a2 + a3, v3 = a3;

  red[o]  = v0 + v1 + v2 + v3;
  red2[o] = v0*v0 + v1*v1 + v2*v2 + v3*v3;
  __syncthreads();
  if (o < 32) {
    float s = red[o] + red[o+32] + red[o+64];
    float q = red2[o] + red2[o+32] + red2[o+64];
    #pragma unroll
    for (int off = 16; off; off >>= 1) {
      s += __shfl_xor(s, off, 32);
      q += __shfl_xor(q, off, 32);
    }
    if (o == 0) { stat[0] = s; stat[1] = q; }
  }
  __syncthreads();
  float mu  = stat[0] * (1.f/384.f);
  float var = stat[1] * (1.f/384.f) - mu*mu;
  float rs  = rsqrtf(var + 1e-5f);

  size_t base = ((size_t)b*LSEQ + H*64 + W) * CDIM;
  float vv[4] = {v0, v1, v2, v3};
  #pragma unroll
  for (int k = 0; k < 4; ++k) {
    int ch = k*96 + o;
    float yv = (vv[k] - mu) * rs * lnw[ch] + lnb[ch];
    y32[base + ch] = yv;
    ybf[base + ch] = (__bf16)yv;
  }
}

// ---------------- kernel 2: QKV projection (WMMA bf16) ----------------
// 4 waves/block, each wave = one 16-position M tile; N=48 (3 tiles), K=384 (12 chunks)
__global__ __launch_bounds__(128)
void qkv_kernel(const __bf16* __restrict__ ybf, const __bf16* __restrict__ wqkv,
                __bf16* __restrict__ qo, __bf16* __restrict__ ko, __bf16* __restrict__ vo) {
  const int wv_  = threadIdx.x >> 5;
  const int lane = threadIdx.x & 31;
  const int half = lane >> 4, n0 = lane & 15;
  const int b = blockIdx.y;
  const int mtile = blockIdx.x*64 + wv_*16;
  const __bf16* yrow = ybf + ((size_t)b*LSEQ + mtile + n0) * CDIM;

  for (int wsel = 0; wsel < 3; ++wsel) {
    const __bf16* wp = wqkv + (size_t)wsel * HDIM * CDIM;
    v8f c0 = vzero(), c1 = vzero(), c2 = vzero();
    for (int kc = 0; kc < 12; ++kc) {
      int kb = half*8 + kc*32;
      V16 ua;
      ua.q[0] = *(const uint4*)(yrow + kb);
      ua.q[1] = *(const uint4*)(yrow + kb + 16);
      int seg = half*8 + kc*32;
      V16 b0, b1, b2;
      b0.q[0] = *(const uint4*)(wp + (size_t)(0*16 + n0)*CDIM + seg);
      b0.q[1] = *(const uint4*)(wp + (size_t)(0*16 + n0)*CDIM + seg + 16);
      b1.q[0] = *(const uint4*)(wp + (size_t)(1*16 + n0)*CDIM + seg);
      b1.q[1] = *(const uint4*)(wp + (size_t)(1*16 + n0)*CDIM + seg + 16);
      b2.q[0] = *(const uint4*)(wp + (size_t)(2*16 + n0)*CDIM + seg);
      b2.q[1] = *(const uint4*)(wp + (size_t)(2*16 + n0)*CDIM + seg + 16);
      c0 = wmma_bf16(ua.v, b0.v, c0);
      c1 = wmma_bf16(ua.v, b1.v, c1);
      c2 = wmma_bf16(ua.v, b2.v, c2);
    }
    if (wsel < 2) {
      __bf16* dst = (wsel == 0) ? qo : ko;
      #pragma unroll
      for (int r = 0; r < 8; ++r) {
        size_t row = (size_t)b*LSEQ + mtile + r + 8*half;
        dst[row*DPAD +  0 + n0] = (__bf16)c0[r];
        dst[row*DPAD + 16 + n0] = (__bf16)c1[r];
        dst[row*DPAD + 32 + n0] = (__bf16)c2[r];
        dst[row*DPAD + 48 + n0] = (__bf16)0.f;   // zero pad channels 48..63
      }
    } else {
      #pragma unroll
      for (int r = 0; r < 8; ++r) {
        int pos = mtile + r + 8*half;
        vo[((size_t)b*HDIM +  0 + n0)*LSEQ + pos] = (__bf16)c0[r];
        vo[((size_t)b*HDIM + 16 + n0)*LSEQ + pos] = (__bf16)c1[r];
        vo[((size_t)b*HDIM + 32 + n0)*LSEQ + pos] = (__bf16)c2[r];
      }
    }
  }
}

// ---------------- kernel 3: flash attention (WMMA bf16) ----------------
// block = 4 waves = 64 queries; key tiles of 64 staged in LDS; online softmax
__global__ __launch_bounds__(128)
void attn_kernel(const __bf16* __restrict__ qg, const __bf16* __restrict__ kg,
                 const __bf16* __restrict__ vg, float* __restrict__ aout) {
  const int tid = threadIdx.x;
  const int wv_  = tid >> 5;
  const int lane = tid & 31;
  const int half = lane >> 4, n0 = lane & 15;
  const int b = blockIdx.y;
  const int qtile = blockIdx.x*64 + wv_*16;

  __shared__ __bf16 Kt[64*DPAD];       // (key, ch)  8KB
  __shared__ __bf16 Vt[HDIM*64];       // (ch, key)  6KB
  __shared__ __bf16 Pt[4][16*64];      // per-wave P (m, key)  8KB

  // Q A-operands for channel chunks 0..31, 32..63 (kept in registers)
  const __bf16* qrow = qg + ((size_t)b*LSEQ + qtile + n0) * DPAD;
  V16 qa0, qa1;
  qa0.q[0] = *(const uint4*)(qrow + half*8);
  qa0.q[1] = *(const uint4*)(qrow + half*8 + 16);
  qa1.q[0] = *(const uint4*)(qrow + half*8 + 32);
  qa1.q[1] = *(const uint4*)(qrow + half*8 + 48);

  float mrun[8], lrun[8];
  #pragma unroll
  for (int r = 0; r < 8; ++r) { mrun[r] = -1e30f; lrun[r] = 0.f; }
  v8f o0 = vzero(), o1 = vzero(), o2 = vzero();

  for (int k0 = 0; k0 < LSEQ; k0 += 64) {
    __syncthreads();
    {   // K tile: rows k0..k0+63 are contiguous in memory (pos-major, stride 64)
      const uint4* src = (const uint4*)(kg + ((size_t)b*LSEQ + k0) * DPAD);
      uint4* d = (uint4*)Kt;
      for (int u = tid; u < 64*DPAD/8; u += 128) d[u] = src[u];
    }
    {   // V tile: 48 rows of 64 keys (ch-major)
      for (int u = tid; u < HDIM*8; u += 128) {
        int c = u >> 3, off = (u & 7) * 8;
        ((uint4*)Vt)[u] = *(const uint4*)(vg + ((size_t)b*HDIM + c)*LSEQ + k0 + off);
      }
    }
    __syncthreads();

    // S = Q K^T  (16 queries x 64 keys), K-dim = channels (0..63, zero-padded)
    v8f s[4];
    #pragma unroll
    for (int nk = 0; nk < 4; ++nk) {
      v8f c = vzero();
      int key = nk*16 + n0;
      V16 ub;
      ub.q[0] = *(const uint4*)(&Kt[key*DPAD + half*8]);
      ub.q[1] = *(const uint4*)(&Kt[key*DPAD + half*8 + 16]);
      c = wmma_bf16(qa0.v, ub.v, c);
      ub.q[0] = *(const uint4*)(&Kt[key*DPAD + 32 + half*8]);
      ub.q[1] = *(const uint4*)(&Kt[key*DPAD + 32 + half*8 + 16]);
      c = wmma_bf16(qa1.v, ub.v, c);
      s[nk] = c;
    }

    // online softmax: per-lane row stats (row = r + 8*half), reduce over N lanes
    float mnew[8], scale[8];
    #pragma unroll
    for (int r = 0; r < 8; ++r) {
      float x = fmaxf(fmaxf(s[0][r], s[1][r]), fmaxf(s[2][r], s[3][r]));
      #pragma unroll
      for (int off = 8; off; off >>= 1) x = fmaxf(x, __shfl_xor(x, off, 32));
      mnew[r]  = fmaxf(mrun[r], x);
      scale[r] = __expf(mrun[r] - mnew[r]);
      mrun[r]  = mnew[r];
    }
    #pragma unroll
    for (int r = 0; r < 8; ++r) {
      o0[r] *= scale[r]; o1[r] *= scale[r]; o2[r] *= scale[r];
      lrun[r] *= scale[r];
    }
    #pragma unroll
    for (int r = 0; r < 8; ++r) {
      float ps = 0.f;
      int m = r + 8*half;
      #pragma unroll
      for (int t = 0; t < 4; ++t) {
        float p = __expf(s[t][r] - mnew[r]);
        ps += p;
        Pt[wv_][m*64 + t*16 + n0] = (__bf16)p;
      }
      #pragma unroll
      for (int off = 8; off; off >>= 1) ps += __shfl_xor(ps, off, 32);
      lrun[r] += ps;
    }

    // O += P V^T  (K-dim = keys 0..63)
    const __bf16* pw = Pt[wv_];
    #pragma unroll
    for (int kk = 0; kk < 2; ++kk) {
      int kb = half*8 + kk*32;
      V16 pa;
      pa.q[0] = *(const uint4*)(&pw[n0*64 + kb]);
      pa.q[1] = *(const uint4*)(&pw[n0*64 + kb + 16]);
      int seg = half*8 + kk*32;
      V16 vb;
      vb.q[0] = *(const uint4*)(&Vt[( 0 + n0)*64 + seg]);
      vb.q[1] = *(const uint4*)(&Vt[( 0 + n0)*64 + seg + 16]);
      o0 = wmma_bf16(pa.v, vb.v, o0);
      vb.q[0] = *(const uint4*)(&Vt[(16 + n0)*64 + seg]);
      vb.q[1] = *(const uint4*)(&Vt[(16 + n0)*64 + seg + 16]);
      o1 = wmma_bf16(pa.v, vb.v, o1);
      vb.q[0] = *(const uint4*)(&Vt[(32 + n0)*64 + seg]);
      vb.q[1] = *(const uint4*)(&Vt[(32 + n0)*64 + seg + 16]);
      o2 = wmma_bf16(pa.v, vb.v, o2);
    }
  }

  #pragma unroll
  for (int r = 0; r < 8; ++r) {
    float inv = 1.f / lrun[r];
    size_t row = (size_t)b*LSEQ + qtile + r + 8*half;
    aout[row*HDIM +  0 + n0] = o0[r] * inv;
    aout[row*HDIM + 16 + n0] = o1[r] * inv;
    aout[row*HDIM + 32 + n0] = o2[r] * inv;
  }
}

// ---------------- kernel 4: gate + smooth + anchor ----------------
__global__ __launch_bounds__(96)
void finalize_kernel(const float* __restrict__ y32, const float* __restrict__ aout,
                     const float* __restrict__ wo, const float* __restrict__ wsm,
                     const float* __restrict__ bsm, const float* __restrict__ anchor,
                     float* __restrict__ out) {
  const int W = blockIdx.x, H = blockIdx.y, b = blockIdx.z;
  const int o = threadIdx.x;
  __shared__ float so[48];
  __shared__ float yq[384];
  size_t pos = (size_t)b*LSEQ + H*64 + W;
  if (o < 48) so[o] = aout[pos*HDIM + o];
  __syncthreads();
  const float* yrow = y32 + pos*CDIM;
  #pragma unroll
  for (int k = 0; k < 4; ++k) {
    int ch = k*96 + o;
    float acc = 0.f;
    const float* wr = wo + (size_t)ch*48;
    #pragma unroll 4
    for (int c = 0; c < 48; ++c) acc = fmaf(wr[c], so[c], acc);
    float att = 1.f / (1.f + __expf(-acc));
    float yv = yrow[ch];
    yq[ch] = att*yv + yv;
  }
  __syncthreads();
  float acc = bsm[o];
  const float* wr = wsm + (size_t)o*CDIM;
  #pragma unroll 4
  for (int i = 0; i < CDIM; ++i) acc = fmaf(wr[i], yq[i], acc);
  size_t oi = (((size_t)b*96 + o)*64 + H)*64 + W;
  out[oi] = acc + anchor[oi];
}

extern "C" void kernel_launch(void* const* d_in, const int* in_sizes, int n_in,
                              void* d_out, int out_size, void* d_ws, size_t ws_size,
                              hipStream_t stream) {
  const float* x0     = (const float*)d_in[0];
  const float* x1     = (const float*)d_in[1];
  const float* x2     = (const float*)d_in[2];
  const float* x3     = (const float*)d_in[3];
  const float* anchor = (const float*)d_in[4];
  const float* wm     = (const float*)d_in[5];
  const float* bm     = (const float*)d_in[6];
  const float* we2    = (const float*)d_in[7];
  const float* be2    = (const float*)d_in[8];
  const float* we4    = (const float*)d_in[9];
  const float* be4    = (const float*)d_in[10];
  const float* lnw    = (const float*)d_in[11];
  const float* lnb    = (const float*)d_in[12];
  const float* wq     = (const float*)d_in[13];
  const float* wk     = (const float*)d_in[14];
  const float* wvv    = (const float*)d_in[15];
  const float* wo     = (const float*)d_in[16];
  const float* wsm    = (const float*)d_in[17];
  const float* bsm    = (const float*)d_in[18];

  char* ws = (char*)d_ws;
  float*  y32  = (float*)(ws + OFF_Y32);
  __bf16* ybf  = (__bf16*)(ws + OFF_YBF);
  __bf16* wqkv = (__bf16*)(ws + OFF_WQKV);
  __bf16* qb   = (__bf16*)(ws + OFF_Q);
  __bf16* kb   = (__bf16*)(ws + OFF_K);
  __bf16* vb   = (__bf16*)(ws + OFF_V);
  float*  ao   = (float*)(ws + OFF_AO);

  cvt_w_kernel<<<(HDIM*CDIM + 255)/256, 256, 0, stream>>>(wq, wk, wvv, wqkv);
  fuse_ln_kernel<<<dim3(64, 64, NB), 96, 0, stream>>>(
      x0, x1, x2, x3, wm, bm, we2, be2, we4, be4, lnw, lnb, y32, ybf);
  qkv_kernel<<<dim3(LSEQ/64, NB), 128, 0, stream>>>(ybf, wqkv, qb, kb, vb);
  attn_kernel<<<dim3(LSEQ/64, NB), 128, 0, stream>>>(qb, kb, vb, ao);
  finalize_kernel<<<dim3(64, 64, NB), 96, 0, stream>>>(
      y32, ao, wo, wsm, bsm, anchor, (float*)d_out);
}
